// DConvolutionModule_51891794870780
// MI455X (gfx1250) — matrix operand order repaired and checked by reading, not compiled, not run on hardware
//
#include <hip/hip_runtime.h>
#include <hip/hip_bf16.h>
#include <float.h>

// ---------------------------------------------------------------------------
// DynamicConv module for MI455X (gfx1250, wave32, WMMA, async-to-LDS).
//   y    = GLU(x @ W1^T + b1)                      (B,T,C)
//   wraw = y @ Wl^T + bl                           (B,T,H*K)
//   w    = band-masked softmax_K(wraw)             (B,T,H,K)
//   z    = sum_j w[.,j] * y[t+j-p] + bias          (B,T,C)
//   out  = z @ W2^T + b2                           (B,T,C)
// B=8 T=2048 C=512 H=4 K=31  ->  M = B*T = 16384
// ---------------------------------------------------------------------------

typedef __attribute__((ext_vector_type(16))) _Float16 v16h;
typedef __attribute__((ext_vector_type(8)))  float    v8f;
typedef __attribute__((ext_vector_type(4)))  int      v4i;

#define LDR 40   // LDS row stride in halves (80B: 16B-aligned rows for B128 async)

union Frag16 { v16h v; uint32_t u[8]; };

// ---- async global->LDS 16B copy (ASYNCcnt path), with sync fallback --------
#if defined(__AMDGCN__) && __has_builtin(__builtin_amdgcn_global_load_async_to_lds_b128)
#define HAVE_ASYNC_LDS 1
#else
#define HAVE_ASYNC_LDS 0
#endif

static __device__ inline void async_copy16(const _Float16* g, _Float16* l) {
#if HAVE_ASYNC_LDS
    __builtin_amdgcn_global_load_async_to_lds_b128(
        (__attribute__((address_space(1))) v4i*)(g),
        (__attribute__((address_space(3))) v4i*)(l), 0, 0);
#else
    float4 tmp = *reinterpret_cast<const float4*>(g);
    *reinterpret_cast<float4*>(l) = tmp;
#endif
}

template <int N>
static __device__ inline void wait_async() {
#if defined(__AMDGCN__)
#if __has_builtin(__builtin_amdgcn_s_wait_asynccnt)
    __builtin_amdgcn_s_wait_asynccnt(N);
#else
    asm volatile("s_wait_asynccnt %0" ::"i"(N) : "memory");
#endif
#endif
}

// ---- WMMA fragment gathers from LDS (ISA 7.12.2 layouts) -------------------
// A (16x32 f16): lane l: m=l&15, h=l>>4; VGPR i = K pair {2i+8h} / {16+2(i-4)+8h}
static __device__ inline v16h load_a_frag(const _Float16* s, int row0, int lane) {
    const int m = lane & 15, h = lane >> 4;
    const _Float16* base = s + (row0 + m) * LDR;
    Frag16 f;
#pragma unroll
    for (int i = 0; i < 8; ++i) {
        int kofs = (i < 4) ? (2 * i + 8 * h) : (16 + 2 * (i - 4) + 8 * h);
        f.u[i] = *reinterpret_cast<const uint32_t*>(base + kofs);
    }
    return f.v;
}

// B (32x16 f16): lane l: n=l&15, h=l>>4; VGPR i = K pair {2i+8h} / {16+2(i-4)+8h}
static __device__ inline v16h load_b_frag(const _Float16* s, int col0, int lane) {
    const int n = lane & 15, h = lane >> 4;
    const _Float16* base = s + (col0 + n) * LDR;
    Frag16 f;
#pragma unroll
    for (int i = 0; i < 8; ++i) {
        f.u[i] = *reinterpret_cast<const uint32_t*>(base + ((i < 4) ? (2 * i + 8 * h) : (16 + 2 * (i - 4) + 8 * h)));
    }
    return f.v;
}

// ---------------------------------------------------------------------------
// WMMA GEMM (f16 operands, fp32 accumulate):
//   Y[m,n] = sum_k X[m,k] * W[n,k]  (+ bias, + optional fused GLU)
// Block: 256 threads (8 waves), tile 128(M) x 64(N), K staged 32/step,
// double-buffered LDS fed by async-to-LDS copies.
// W rows must be valid up to the padded block edge (Wl is zero-padded to 128).
// ---------------------------------------------------------------------------
template <bool GLU>
__global__ __launch_bounds__(256)
void wmma_gemm_f16(const _Float16* __restrict__ Xh, const _Float16* __restrict__ Wh,
                   const float* __restrict__ bvec, float* __restrict__ Yf,
                   _Float16* __restrict__ Yh,
                   int Kd, int Nvalid, int halfOff, int ldY) {
    __shared__ _Float16 sX [2][128 * LDR];
    __shared__ _Float16 sWa[2][ 64 * LDR];
    __shared__ _Float16 sWg[2][ 64 * LDR];

    const int tid  = threadIdx.x;
    const int lane = tid & 31;
    const int wave = tid >> 5;
    const int wm   = wave >> 1;   // 0..3
    const int wn   = wave & 1;    // 0..1
    const int m0   = blockIdx.x * 128;
    const int n0   = blockIdx.y * 64;

    // per-thread staging coordinates
    const int xrow0 = tid >> 2;            // X chunk 0: rows 0..63
    const int xrow1 = 64 + (tid >> 2);     // X chunk 1: rows 64..127
    const int wrow  = tid >> 2;            // W chunk : rows 0..63
    const int ck    = (tid & 3) * 8;       // 8-half (16B) column chunk

    constexpr int PER_STAGE = GLU ? 4 : 3; // async instructions per wave per stage

    auto issue_stage = [&](int buf, int k0) {
        async_copy16(&Xh[(size_t)(m0 + xrow0) * Kd + k0 + ck],
                     &sX[buf][xrow0 * LDR + ck]);
        async_copy16(&Xh[(size_t)(m0 + xrow1) * Kd + k0 + ck],
                     &sX[buf][xrow1 * LDR + ck]);
        async_copy16(&Wh[(size_t)(n0 + wrow) * Kd + k0 + ck],
                     &sWa[buf][wrow * LDR + ck]);
        if (GLU)
            async_copy16(&Wh[(size_t)(halfOff + n0 + wrow) * Kd + k0 + ck],
                         &sWg[buf][wrow * LDR + ck]);
    };

    v8f acc[2][2];
    v8f accg[2][2];
    const v8f vzero = {0.f, 0.f, 0.f, 0.f, 0.f, 0.f, 0.f, 0.f};
#pragma unroll
    for (int i = 0; i < 2; ++i)
#pragma unroll
        for (int j = 0; j < 2; ++j) { acc[i][j] = vzero; accg[i][j] = vzero; }

    int buf = 0;
    issue_stage(0, 0);

    for (int k0 = 0; k0 < Kd; k0 += 32) {
        const bool more = (k0 + 32) < Kd;
        if (more) {
            issue_stage(buf ^ 1, k0 + 32);   // prefetch next stage
            wait_async<PER_STAGE>();         // only the prefetch stays in flight
        } else {
            wait_async<0>();
        }
        __syncthreads();                     // LDS writes visible to all waves

        v16h afrag[2];
#pragma unroll
        for (int ms = 0; ms < 2; ++ms)
            afrag[ms] = load_a_frag(sX[buf], wm * 32 + ms * 16, lane);
#pragma unroll
        for (int ns = 0; ns < 2; ++ns) {
            v16h bfrag = load_b_frag(sWa[buf], wn * 32 + ns * 16, lane);
#pragma unroll
            for (int ms = 0; ms < 2; ++ms)
                acc[ms][ns] = __builtin_amdgcn_wmma_f32_16x16x32_f16(
                    false, afrag[ms], false, bfrag, (short)0, acc[ms][ns],
                    false, false);
            if (GLU) {
                v16h gfrag = load_b_frag(sWg[buf], wn * 32 + ns * 16, lane);
#pragma unroll
                for (int ms = 0; ms < 2; ++ms)
                    accg[ms][ns] = __builtin_amdgcn_wmma_f32_16x16x32_f16(
                        false, afrag[ms], false, gfrag, (short)0, accg[ms][ns],
                        false, false);
            }
        }
        __syncthreads();                     // done reading buf before overwrite
        buf ^= 1;
    }

    // ---- epilogue: C/D layout: lane l -> col l&15, rows (l>>4)*8 + r ----
    const int cl = lane & 15;
    const int hh = lane >> 4;
#pragma unroll
    for (int ms = 0; ms < 2; ++ms)
#pragma unroll
        for (int ns = 0; ns < 2; ++ns) {
            int gm = m0 + wm * 32 + ms * 16 + hh * 8;
            int gn = n0 + wn * 32 + ns * 16 + cl;
            if (gn >= Nvalid) continue;
            if (GLU) {
                float ba = bvec[gn];
                float bg = bvec[halfOff + gn];
#pragma unroll
                for (int r = 0; r < 8; ++r) {
                    float a = acc[ms][ns][r] + ba;
                    float g = accg[ms][ns][r] + bg;
                    float yv = a * (1.0f / (1.0f + __expf(-g)));
                    size_t o = (size_t)(gm + r) * ldY + gn;
                    Yf[o] = yv;
                    Yh[o] = (_Float16)yv;    // f16 copy feeds the next GEMM
                }
            } else {
                float bb = bvec[gn];
#pragma unroll
                for (int r = 0; r < 8; ++r)
                    Yf[(size_t)(gm + r) * ldY + gn] = acc[ms][ns][r] + bb;
            }
        }
}

// ---------------------------------------------------------------------------
// fp32 -> f16 pack kernels (one-time operand conversion)
// ---------------------------------------------------------------------------
__global__ __launch_bounds__(256)
void pack_f16_kernel(const float* __restrict__ src, _Float16* __restrict__ dst, int n) {
    int i = blockIdx.x * 256 + threadIdx.x;
    if (i < n) dst[i] = (_Float16)src[i];
}

__global__ __launch_bounds__(256)
void pack_f16_pad_kernel(const float* __restrict__ src, _Float16* __restrict__ dst,
                         int rows, int padRows, int cols) {
    int i = blockIdx.x * 256 + threadIdx.x;
    if (i >= padRows * cols) return;
    int r = i / cols, c = i - r * cols;
    dst[i] = (r < rows) ? (_Float16)src[(size_t)r * cols + c] : (_Float16)0.f;
}

// ---------------------------------------------------------------------------
// Band-masked softmax over K=31 taps, in-place on w (BT, H*K).
// ---------------------------------------------------------------------------
__global__ __launch_bounds__(256)
void band_softmax_kernel(float* __restrict__ w, int BT, int T) {
    int idx = blockIdx.x * 256 + threadIdx.x;   // over BT*H
    if (idx >= BT * 4) return;
    int bt = idx >> 2;
    int h  = idx & 3;
    int t  = bt % T;
    float* base = w + (size_t)bt * 124 + h * 31;

    float vals[31];
    float mx = -FLT_MAX;
#pragma unroll
    for (int j = 0; j < 31; ++j) {
        int src = t + j - 15;
        bool ok = (src >= 0) & (src < T);
        float x = ok ? base[j] : -FLT_MAX;
        vals[j] = x;
        mx = fmaxf(mx, x);
    }
    float s = 0.f;
#pragma unroll
    for (int j = 0; j < 31; ++j) {
        float e = __expf(vals[j] - mx);   // masked taps underflow to 0
        vals[j] = e;
        s += e;
    }
    float inv = 1.f / s;
#pragma unroll
    for (int j = 0; j < 31; ++j) base[j] = vals[j] * inv;
}

// ---------------------------------------------------------------------------
// Banded dynamic conv: z[bt,c..c+3] = (f16)(bias + sum_j w[bt,h,j]*y[b,t+j-p,c..])
// y (~33.5 MB fp32) is L2-resident on the 192 MB L2; gathers hit L2.
// Writes z directly in f16 (its only consumer is the WMMA GEMM2).
// ---------------------------------------------------------------------------
struct __align__(8) Half4 { _Float16 a, b, c, d; };

__global__ __launch_bounds__(256)
void dconv_kernel(const float* __restrict__ y, const float* __restrict__ w,
                  const float* __restrict__ bias, _Float16* __restrict__ zh, int T) {
    int idx = blockIdx.x * 256 + threadIdx.x;   // over BT * (C/4)
    int bt = idx >> 7;
    int c  = (idx & 127) * 4;
    int b  = bt / T;
    int t  = bt % T;
    int h  = c >> 7;                            // 128 channels per head
    const float* wp = w + (size_t)bt * 124 + h * 31;

    float a0 = bias[c + 0], a1 = bias[c + 1], a2 = bias[c + 2], a3 = bias[c + 3];
#pragma unroll
    for (int j = 0; j < 31; ++j) {
        int src = t + j - 15;
        if (src >= 0 && src < T) {
            float wv = wp[j];
            float4 yv = *reinterpret_cast<const float4*>(
                &y[((size_t)b * T + src) * 512 + c]);
            a0 += wv * yv.x; a1 += wv * yv.y; a2 += wv * yv.z; a3 += wv * yv.w;
        }
    }
    Half4 o = {(_Float16)a0, (_Float16)a1, (_Float16)a2, (_Float16)a3};
    *reinterpret_cast<Half4*>(&zh[(size_t)bt * 512 + c]) = o;
}

// ---------------------------------------------------------------------------
extern "C" void kernel_launch(void* const* d_in, const int* in_sizes, int n_in,
                              void* d_out, int out_size, void* d_ws, size_t ws_size,
                              hipStream_t stream) {
    (void)in_sizes; (void)n_in; (void)out_size; (void)ws_size;
    const float* x    = (const float*)d_in[0];   // (B,T,C)       8*2048*512
    const float* W1   = (const float*)d_in[1];   // (2C, C)       1024*512
    const float* b1   = (const float*)d_in[2];   // (2C,)
    const float* Wl   = (const float*)d_in[3];   // (H*K, C)      124*512
    const float* bl   = (const float*)d_in[4];   // (H*K,)
    const float* bias = (const float*)d_in[5];   // (C,)
    const float* W2   = (const float*)d_in[6];   // (C, C)        512*512
    const float* b2   = (const float*)d_in[7];   // (C,)
    float* out = (float*)d_out;                  // (B,T,C) fp32

    const int B = 8, T = 2048, C = 512;
    const int M = B * T;                         // 16384

    // ---- workspace layout: fp32 region first, then f16 region ----
    float* y    = (float*)d_ws;                          // M*C        fp32
    float* wbuf = y + (size_t)M * C;                     // M*124      fp32
    _Float16* xh  = (_Float16*)(wbuf + (size_t)M * 124); // M*C        f16
    _Float16* yh  = xh  + (size_t)M * C;                 // M*C        f16
    _Float16* zh  = yh  + (size_t)M * C;                 // M*C        f16
    _Float16* w1h = zh  + (size_t)M * C;                 // 1024*512   f16
    _Float16* wlh = w1h + (size_t)1024 * 512;            // 128*512    f16 (padded)
    _Float16* w2h = wlh + (size_t)128 * 512;             // 512*512    f16

    // 0) one-time f16 packing of GEMM operands
    pack_f16_kernel<<<(M * C) / 256, 256, 0, stream>>>(x, xh, M * C);
    pack_f16_kernel<<<(1024 * 512) / 256, 256, 0, stream>>>(W1, w1h, 1024 * 512);
    pack_f16_pad_kernel<<<(128 * 512) / 256, 256, 0, stream>>>(Wl, wlh, 124, 128, 512);
    pack_f16_kernel<<<(512 * 512) / 256, 256, 0, stream>>>(W2, w2h, 512 * 512);

    // 1) y = GLU(x @ W1^T + b1)   (writes fp32 y for conv + f16 yh for GEMM-l)
    wmma_gemm_f16<true><<<dim3(M / 128, C / 64), 256, 0, stream>>>(
        xh, w1h, b1, y, yh, /*Kd=*/C, /*Nvalid=*/C, /*halfOff=*/C, /*ldY=*/C);

    // 2) wraw = y @ Wl^T + bl     (N=124 valid, W zero-padded to 128 rows)
    wmma_gemm_f16<false><<<dim3(M / 128, 2), 256, 0, stream>>>(
        yh, wlh, bl, wbuf, nullptr, /*Kd=*/C, /*Nvalid=*/124, /*halfOff=*/0, /*ldY=*/124);

    // 3) band-masked softmax over K=31 taps (in-place)
    band_softmax_kernel<<<(M * 4) / 256, 256, 0, stream>>>(wbuf, M, T);

    // 4) z = dynconv(y, w) + bias  (f16 output feeds GEMM2)
    dconv_kernel<<<(M * (C / 4)) / 256, 256, 0, stream>>>(y, wbuf, bias, zh, T);

    // 5) out = z @ W2^T + b2
    wmma_gemm_f16<false><<<dim3(M / 128, C / 64), 256, 0, stream>>>(
        zh, w2h, b2, out, nullptr, /*Kd=*/C, /*Nvalid=*/C, /*halfOff=*/0, /*ldY=*/C);
}
